// MultiHeadAttention_65953517797857
// MI455X (gfx1250) — matrix operand order repaired and checked by reading, not compiled, not run on hardware
//
#include <hip/hip_runtime.h>
#include <math.h>

// ---------- problem constants ----------
#define HH   8
#define DIN  512
#define HD   64
#define BB   16
#define NSEQ 1024
#define ROWS (BB * NSEQ)   // 16384 rows of q (b,n flattened)

// ---------- WMMA types ----------
typedef __attribute__((ext_vector_type(16))) __bf16 v16bf;
typedef __attribute__((ext_vector_type(8)))  __bf16 v8bf;
typedef __attribute__((ext_vector_type(8)))  float  v8f;

static __device__ __forceinline__ __bf16 to_bf16(float f) {
    unsigned u = __builtin_bit_cast(unsigned, f);
    unsigned r = u + 0x7FFFu + ((u >> 16) & 1u);     // round-to-nearest-even
    unsigned short h = (unsigned short)(r >> 16);
    return __builtin_bit_cast(__bf16, h);
}

static __device__ __forceinline__ v8f wmma_bf16(v16bf a, v16bf b, v8f c) {
    return __builtin_amdgcn_wmma_f32_16x16x32_bf16(false, a, false, b,
                                                   (short)0, c, false, false);
}

// ---------- CDNA5 async global->LDS copy (ASYNCcnt tracked) ----------
static __device__ __forceinline__ void async_load_b128(unsigned lds_byte_off,
                                                       const void* gaddr) {
    asm volatile("global_load_async_to_lds_b128 %0, %1, off"
                 :: "v"(lds_byte_off), "v"(gaddr)
                 : "memory");
}
static __device__ __forceinline__ void wait_async0() {
#if __has_builtin(__builtin_amdgcn_s_wait_asynccnt)
    __builtin_amdgcn_s_wait_asynccnt(0);
#else
    asm volatile("s_wait_asynccnt 0x0" ::: "memory");
#endif
}

// A fragment: 16x32 (MxK), row-major source, leading dim `ld` (elements).
static __device__ __forceinline__ v16bf load_fragA(const __bf16* __restrict__ base,
                                                   int ld, int row0, int kbase, int lane) {
    int m    = lane & 15;
    int half = (lane >> 4) & 1;
    const __bf16* p = base + (size_t)(row0 + m) * ld + kbase + half * 8;
    v8bf lo = *(const v8bf*)p;
    v8bf hi = *(const v8bf*)(p + 16);
    v16bf r;
#pragma unroll
    for (int i = 0; i < 8; ++i) { r[i] = lo[i]; r[i + 8] = hi[i]; }
    return r;
}

// B fragment: 32x16 (KxN). Source "transposed": row = output column, K contiguous.
static __device__ __forceinline__ v16bf load_fragB(const __bf16* __restrict__ base,
                                                   int ld, int col0, int kbase, int lane) {
    int n    = lane & 15;
    int half = (lane >> 4) & 1;
    const __bf16* p = base + (size_t)(col0 + n) * ld + kbase + half * 8;
    v8bf lo = *(const v8bf*)p;
    v8bf hi = *(const v8bf*)(p + 16);
    v16bf r;
#pragma unroll
    for (int i = 0; i < 8; ++i) { r[i] = lo[i]; r[i + 8] = hi[i]; }
    return r;
}

// ---------- kernel 1: q (f32) -> bf16 ----------
__global__ __launch_bounds__(256)
void k_cvt_q(const float* __restrict__ q, __bf16* __restrict__ qbf, int n) {
    int i = blockIdx.x * blockDim.x + threadIdx.x;
    int stride = gridDim.x * blockDim.x;
    for (; i < n; i += stride) qbf[i] = to_bf16(q[i]);
}

// ---------- kernel 2: transpose+convert all weights ----------
__global__ __launch_bounds__(256)
void k_cvt_w(const float* __restrict__ Wq, const float* __restrict__ Wk,
             const float* __restrict__ Wv, const float* __restrict__ Wo,
             __bf16* __restrict__ WqT, __bf16* __restrict__ WkT,
             __bf16* __restrict__ WvT, __bf16* __restrict__ WoT) {
    int i = blockIdx.x * blockDim.x + threadIdx.x;
    if (i >= HH * DIN * HD) return;                    // 262144
    int h   = i / (DIN * HD);
    int rem = i - h * (DIN * HD);
    int d   = rem / HD;
    int k   = rem - d * HD;
    size_t src = (size_t)h * DIN * HD + (size_t)d * HD + k;
    size_t dst = (size_t)h * HD * DIN + (size_t)k * DIN + d;
    WqT[dst] = to_bf16(Wq[src]);
    WkT[dst] = to_bf16(Wk[src]);
    WvT[dst] = to_bf16(Wv[src]);
    int e = i >> 9, hv = i & 511;                      // 512x512 transpose view
    WoT[(size_t)e * 512 + hv] = to_bf16(Wo[(size_t)hv * 512 + e]);
}

// ---------- kernel 3: QKV projections ----------
// grid (64, 8, 3), block 256 (8 waves). Wave tile: 32 rows x 64 cols.
// Static ping-pong double buffering (no register rotation moves).
__global__ __launch_bounds__(256)
void k_proj(const __bf16* __restrict__ qbf,
            const __bf16* __restrict__ WqT, const __bf16* __restrict__ WkT,
            const __bf16* __restrict__ WvT,
            __bf16* __restrict__ Qh, __bf16* __restrict__ Kh,
            __bf16* __restrict__ VT) {
    int lane  = threadIdx.x & 31;
    int wave  = threadIdx.x >> 5;
    int mtile = blockIdx.x * 8 + wave;     // 0..511 (32-row tiles)
    int h     = blockIdx.y;
    int proj  = blockIdx.z;                // 0=Q 1=K 2=V
    const __bf16* WT = (proj == 0) ? WqT : (proj == 1) ? WkT : WvT;
    WT += (size_t)h * HD * DIN;
    int m0 = mtile * 32;

    v8f accA[4] = {}, accB[4] = {};
    v16bf a0[2], a1[2], bfr[2][4];
    a0[0] = load_fragA(qbf, DIN, m0,      0, lane);
    a1[0] = load_fragA(qbf, DIN, m0 + 16, 0, lane);
#pragma unroll
    for (int t = 0; t < 4; ++t) bfr[0][t] = load_fragB(WT, DIN, t * 16, 0, lane);

    for (int kb = 0; kb < DIN; kb += 64) {             // 8 iters, 2 chunks each
        // stage chunk kb+32 into set 1
        a0[1] = load_fragA(qbf, DIN, m0,      kb + 32, lane);
        a1[1] = load_fragA(qbf, DIN, m0 + 16, kb + 32, lane);
#pragma unroll
        for (int t = 0; t < 4; ++t) bfr[1][t] = load_fragB(WT, DIN, t * 16, kb + 32, lane);
        // compute chunk kb from set 0
#pragma unroll
        for (int t = 0; t < 4; ++t) {
            accA[t] = wmma_bf16(a0[0], bfr[0][t], accA[t]);
            accB[t] = wmma_bf16(a1[0], bfr[0][t], accB[t]);
        }
        // stage chunk kb+64 into set 0 (dummy reload on last iter)
        int kn = (kb + 64 < DIN) ? kb + 64 : 0;
        a0[0] = load_fragA(qbf, DIN, m0,      kn, lane);
        a1[0] = load_fragA(qbf, DIN, m0 + 16, kn, lane);
#pragma unroll
        for (int t = 0; t < 4; ++t) bfr[0][t] = load_fragB(WT, DIN, t * 16, kn, lane);
        // compute chunk kb+32 from set 1
#pragma unroll
        for (int t = 0; t < 4; ++t) {
            accA[t] = wmma_bf16(a0[1], bfr[1][t], accA[t]);
            accB[t] = wmma_bf16(a1[1], bfr[1][t], accB[t]);
        }
    }

    int n  = lane & 15;
    int mo = (lane >> 4) * 8;
    if (proj == 2) {                       // transposed store: VT[h][b][d][g]
        int b  = m0 >> 10;
        int g0 = m0 & 1023;
        __bf16* dst = VT + (size_t)(h * BB + b) * HD * NSEQ;
#pragma unroll
        for (int rr = 0; rr < 8; ++rr) {
            int gA = g0 + rr + mo;
            int gB = gA + 16;
#pragma unroll
            for (int t = 0; t < 4; ++t) {
                dst[(size_t)(t * 16 + n) * NSEQ + gA] = to_bf16(accA[t][rr]);
                dst[(size_t)(t * 16 + n) * NSEQ + gB] = to_bf16(accB[t][rr]);
            }
        }
    } else {
        __bf16* dst = ((proj == 0) ? Qh : Kh) + ((size_t)h * ROWS + m0) * HD;
#pragma unroll
        for (int rr = 0; rr < 8; ++rr) {
            size_t rA = (size_t)(rr + mo) * HD;
            size_t rB = (size_t)(rr + mo + 16) * HD;
#pragma unroll
            for (int t = 0; t < 4; ++t) {
                dst[rA + t * 16 + n] = to_bf16(accA[t][rr]);
                dst[rB + t * 16 + n] = to_bf16(accB[t][rr]);
            }
        }
    }
}

// ---------- kernel 4: attention core ----------
// grid (16, B, H), block 128 (4 waves); LDS = 256KB scores + 2x8KB stage buffers.
// K/V chunks double-buffered via global_load_async_to_lds_b128: DMA for chunk
// c+1 overlaps WMMA on chunk c; first V chunk DMA overlaps the softmax phase.
#define SC_BYTES (4 * 16 * NSEQ * 4)   // 262144
#define STG_OFF  SC_BYTES
#define STG_SZ   8192
__global__ __launch_bounds__(128)
void k_attn(const __bf16* __restrict__ Qh, const __bf16* __restrict__ Kh,
            const __bf16* __restrict__ VT, __bf16* __restrict__ headsC) {
    extern __shared__ char smem[];
    int tid  = threadIdx.x;
    int lane = tid & 31;
    int wave = tid >> 5;
    int b = blockIdx.y, h = blockIdx.z;
    int q0 = (blockIdx.x * 4 + wave) * 16;

    float* sc = (float*)(smem + (size_t)wave * 16 * NSEQ * 4);

    const __bf16* Qb = Qh + ((size_t)h * ROWS + (size_t)b * NSEQ) * HD;
    const __bf16* Kb = Kh + ((size_t)h * ROWS + (size_t)b * NSEQ) * HD;
    const __bf16* Vb = VT + (size_t)(h * BB + b) * HD * NSEQ;

    v16bf aq0 = load_fragA(Qb, HD, q0,  0, lane);
    v16bf aq1 = load_fragA(Qb, HD, q0, 32, lane);

    int n  = lane & 15;
    int mo = (lane >> 4) * 8;
    int cr   = tid >> 1;                 // staging row handled by this thread
    int part = tid & 1;                  // which 64B half of the 128B row

    // stage 64 contiguous rows of 128B starting at row base (K rows / VT rows)
    auto issueK = [&](int c0, unsigned boff) {
        const char* g = (const char*)(Kb + (size_t)(c0 + cr) * HD) + part * 64;
        unsigned   l  = boff + cr * 128 + part * 64;
        async_load_b128(l,      g);
        async_load_b128(l + 16, g + 16);
        async_load_b128(l + 32, g + 32);
        async_load_b128(l + 48, g + 48);
    };
    auto issueV = [&](int c0, unsigned boff) {
        const char* g = (const char*)(Vb + (size_t)cr * NSEQ + c0) + part * 64;
        unsigned   l  = boff + cr * 128 + part * 64;
        async_load_b128(l,      g);
        async_load_b128(l + 16, g + 16);
        async_load_b128(l + 32, g + 32);
        async_load_b128(l + 48, g + 48);
    };

    // ---- scores = 0.125 * Q K^T, self-masked; K double-buffered in LDS ----
    issueK(0, STG_OFF);
    for (int c0 = 0; c0 < NSEQ; c0 += 64) {
        int buf = (c0 >> 6) & 1;
        wait_async0();                                  // chunk c0 landed
        __syncthreads();                                // visible block-wide
        if (c0 + 64 < NSEQ) issueK(c0 + 64, STG_OFF + (buf ^ 1) * STG_SZ);
        const __bf16* stg = (const __bf16*)(smem + STG_OFF + buf * STG_SZ);
        for (int g0 = c0; g0 < c0 + 64; g0 += 16) {
            v16bf bk0 = load_fragB(stg, HD, g0 - c0,  0, lane);
            v16bf bk1 = load_fragB(stg, HD, g0 - c0, 32, lane);
            v8f s = {};
            s = wmma_bf16(aq0, bk0, s);
            s = wmma_bf16(aq1, bk1, s);
#pragma unroll
            for (int rr = 0; rr < 8; ++rr) {
                int m = rr + mo;
                float v = 0.125f * s[rr];
                if (q0 + m == g0 + n) v = -3.0e38f;    // self-mask
                sc[(size_t)m * NSEQ + g0 + n] = v;
            }
        }
    }
    // first V chunk DMA overlaps the softmax phase (buf0 free: last readers done)
    issueV(0, STG_OFF);
    __syncthreads();

    // ---- softmax per row; probs packed bf16 in place (write idx <= read idx) ----
    if (lane < 16) {
        float* srow = sc + (size_t)lane * NSEQ;
        float mx = -3.4e38f;
        for (int j = 0; j < NSEQ; ++j) mx = fmaxf(mx, srow[j]);
        float sum = 0.f;
        for (int j = 0; j < NSEQ; ++j) sum += __expf(srow[j] - mx);
        float inv = 1.0f / sum;
        for (int j = 0; j < NSEQ; j += 2) {
            float p0 = __expf(srow[j]     - mx) * inv;
            float p1 = __expf(srow[j + 1] - mx) * inv;
            unsigned u = (unsigned)__builtin_bit_cast(unsigned short, to_bf16(p0))
                       | ((unsigned)__builtin_bit_cast(unsigned short, to_bf16(p1)) << 16);
            srow[j >> 1] = __builtin_bit_cast(float, u);
        }
    }
    __syncthreads();

    // ---- heads = probs x V; V double-buffered in LDS ----
    const __bf16* pr = (const __bf16*)sc;              // row stride 2048 bf16
    v8f acc[4] = {};
    for (int c0 = 0; c0 < NSEQ; c0 += 64) {
        int buf = (c0 >> 6) & 1;
        wait_async0();
        __syncthreads();
        if (c0 + 64 < NSEQ) issueV(c0 + 64, STG_OFF + (buf ^ 1) * STG_SZ);
        const __bf16* stg = (const __bf16*)(smem + STG_OFF + buf * STG_SZ);
        for (int g0 = c0; g0 < c0 + 64; g0 += 32) {
            v16bf a = load_fragA(pr, 2048, 0, g0, lane);
#pragma unroll
            for (int t = 0; t < 4; ++t) {
                v16bf bv = load_fragB(stg, 64, t * 16, g0 - c0, lane);
                acc[t] = wmma_bf16(a, bv, acc[t]);
            }
        }
    }

    __bf16* dst = headsC + ((size_t)b * NSEQ + q0) * (HH * HD) + h * HD;
#pragma unroll
    for (int rr = 0; rr < 8; ++rr) {
        size_t r = (size_t)(rr + mo) * (HH * HD);
#pragma unroll
        for (int t = 0; t < 4; ++t)
            dst[r + t * 16 + n] = to_bf16(acc[t][rr]);
    }
}

// ---------- kernel 5: output projection ----------
// out (16384x512 f32) = headsC (16384x512 bf16) @ WoT^T. 32x64 tiles, ping-pong.
__global__ __launch_bounds__(256)
void k_out(const __bf16* __restrict__ headsC, const __bf16* __restrict__ WoT,
           float* __restrict__ out) {
    int lane = threadIdx.x & 31;
    int wid  = blockIdx.x * 8 + (threadIdx.x >> 5);    // 0..4095
    int m0   = (wid >> 3) * 32;
    int e0   = (wid & 7) * 64;

    v8f accA[4] = {}, accB[4] = {};
    v16bf a0[2], a1[2], bfr[2][4];
    a0[0] = load_fragA(headsC, 512, m0,      0, lane);
    a1[0] = load_fragA(headsC, 512, m0 + 16, 0, lane);
#pragma unroll
    for (int t = 0; t < 4; ++t) bfr[0][t] = load_fragB(WoT, 512, e0 + t * 16, 0, lane);

    for (int kb = 0; kb < 512; kb += 64) {
        a0[1] = load_fragA(headsC, 512, m0,      kb + 32, lane);
        a1[1] = load_fragA(headsC, 512, m0 + 16, kb + 32, lane);
#pragma unroll
        for (int t = 0; t < 4; ++t) bfr[1][t] = load_fragB(WoT, 512, e0 + t * 16, kb + 32, lane);
#pragma unroll
        for (int t = 0; t < 4; ++t) {
            accA[t] = wmma_bf16(a0[0], bfr[0][t], accA[t]);
            accB[t] = wmma_bf16(a1[0], bfr[0][t], accB[t]);
        }
        int kn = (kb + 64 < 512) ? kb + 64 : 0;
        a0[0] = load_fragA(headsC, 512, m0,      kn, lane);
        a1[0] = load_fragA(headsC, 512, m0 + 16, kn, lane);
#pragma unroll
        for (int t = 0; t < 4; ++t) bfr[0][t] = load_fragB(WoT, 512, e0 + t * 16, kn, lane);
#pragma unroll
        for (int t = 0; t < 4; ++t) {
            accA[t] = wmma_bf16(a0[1], bfr[1][t], accA[t]);
            accB[t] = wmma_bf16(a1[1], bfr[1][t], accB[t]);
        }
    }

    int n  = lane & 15;
    int mo = (lane >> 4) * 8;
#pragma unroll
    for (int rr = 0; rr < 8; ++rr) {
        size_t rA = (size_t)(m0 + rr + mo) * 512 + e0;
        size_t rB = (size_t)(m0 + rr + mo + 16) * 512 + e0;
#pragma unroll
        for (int t = 0; t < 4; ++t) {
            out[rA + t * 16 + n] = accA[t][rr];
            out[rB + t * 16 + n] = accB[t][rr];
        }
    }
}

// ---------- host launcher ----------
extern "C" void kernel_launch(void* const* d_in, const int* in_sizes, int n_in,
                              void* d_out, int out_size, void* d_ws, size_t ws_size,
                              hipStream_t stream) {
    const float* q  = (const float*)d_in[0];
    const float* Wq = (const float*)d_in[1];
    const float* Wk = (const float*)d_in[2];
    const float* Wv = (const float*)d_in[3];
    const float* Wo = (const float*)d_in[4];
    float* out = (float*)d_out;

    size_t off = 0;
    char* base = (char*)d_ws;
    auto take = [&](size_t bytes) -> char* {
        char* p = base + off;
        off += (bytes + 255) & ~(size_t)255;
        return p;
    };
    __bf16* qbf = (__bf16*)take((size_t)ROWS * DIN * 2);        // 16 MB
    __bf16* WqT = (__bf16*)take((size_t)HH * HD * DIN * 2);     // 512 KB
    __bf16* WkT = (__bf16*)take((size_t)HH * HD * DIN * 2);
    __bf16* WvT = (__bf16*)take((size_t)HH * HD * DIN * 2);
    __bf16* WoT = (__bf16*)take((size_t)512 * 512 * 2);
    __bf16* Qh  = (__bf16*)take((size_t)HH * ROWS * HD * 2);    // 16 MB
    __bf16* Kh  = (__bf16*)take((size_t)HH * ROWS * HD * 2);    // 16 MB
    __bf16* VT  = (__bf16*)take((size_t)HH * ROWS * HD * 2);    // 16 MB
    __bf16* hC  = (__bf16*)take((size_t)ROWS * HH * HD * 2);    // 16 MB

    k_cvt_q<<<2048, 256, 0, stream>>>(q, qbf, ROWS * DIN);
    k_cvt_w<<<1024, 256, 0, stream>>>(Wq, Wk, Wv, Wo, WqT, WkT, WvT, WoT);
    k_proj<<<dim3(64, 8, 3), 256, 0, stream>>>(qbf, WqT, WkT, WvT, Qh, Kh, VT);
    k_attn<<<dim3(16, BB, HH), 128, SC_BYTES + 2 * STG_SZ, stream>>>(Qh, Kh, VT, hC);
    k_out<<<512, 256, 0, stream>>>(hC, WoT, out);
}
